// Attention_28020366639185
// MI455X (gfx1250) — matrix operand order repaired and checked by reading, not compiled, not run on hardware
//
#include <hip/hip_runtime.h>
#include <hip/hip_bf16.h>

typedef __attribute__((ext_vector_type(16))) _Float16 v16h;
typedef __attribute__((ext_vector_type(8)))  _Float16 v8h;
typedef __attribute__((ext_vector_type(8)))  float    v8f;

#define DIM   1024
#define NH    16
#define HD    64
#define SEQ   2048
#define BATCH 2
#define ROWS  (BATCH * SEQ)       // 4096
#define N_QKV (3 * NH * HD)       // 3072

// ---------------- workspace layout (bytes) ----------------
#define XN_OFF     ((size_t)0)                          //  8 MB  xn f16 (4096x1024)
#define WQKVT_OFF  ((size_t)8  * 1024 * 1024)           //  6 MB  w_qkv^T f16 (3072x1024)
#define WOUTT_OFF  ((size_t)14 * 1024 * 1024)           //  2 MB  w_out^T f16 (1024x1024)
#define QB_OFF     ((size_t)16 * 1024 * 1024)           //  8 MB  Q  (b,h,n,d) f16
#define KB_OFF     ((size_t)24 * 1024 * 1024)           //  8 MB  K  (b,h,n,d) f16
#define VTB_OFF    ((size_t)32 * 1024 * 1024)           //  8 MB  V^T (b,h,d,n) f16
#define ATTN_OFF   ((size_t)40 * 1024 * 1024)           //  8 MB  attn (b*n, h*hd) f16

// ---------------- WMMA helpers ----------------
__device__ __forceinline__ v8f wmma_f16(v16h a, v16h b, v8f c) {
  return __builtin_amdgcn_wmma_f32_16x16x32_f16(
      /*neg_a=*/false, a, /*neg_b=*/false, b,
      /*c_mod=*/(short)0, c, /*reuse_a=*/false, /*reuse_b=*/false);
}

__device__ __forceinline__ v16h combine16(v8h lo, v8h hi) {
  v16h r;
#pragma unroll
  for (int i = 0; i < 8; ++i) { r[i] = lo[i]; r[8 + i] = hi[i]; }
  return r;
}

// A fragment (16x32, MxK) from row-major [ld = lda], rows row0..row0+15, K = k0..k0+31.
// ISA layout: lane m=L&15, half=L>>4; elems 0..7 -> K = half*8 + j ; 8..15 -> K = 16 + half*8 + j.
__device__ __forceinline__ v16h load_a_frag(const _Float16* __restrict__ base, int ld,
                                            int row0, int k0, int lane) {
  int m = lane & 15, half = lane >> 4;
  const _Float16* p = base + (size_t)(row0 + m) * ld + k0 + half * 8;
  v8h lo = *(const v8h*)p;
  v8h hi = *(const v8h*)(p + 16);
  return combine16(lo, hi);
}

// B fragment (32x16, KxN) taken from a TRANSPOSED operand Bt (N x K, row-major, ld = ldb):
// lane n=L&15 holds B[k, col0+n] = Bt[col0+n, k]; elems 0..7 -> k = k0 + half*8 + j ; 8..15 -> k = k0 + 16 + half*8 + j.
__device__ __forceinline__ v16h load_bt_frag(const _Float16* __restrict__ base, int ld,
                                             int col0, int k0, int lane) {
  int n = lane & 15, kb = (lane >> 4) * 8;
  const _Float16* p = base + (size_t)(col0 + n) * ld + k0 + kb;
  v8h lo = *(const v8h*)p;
  v8h hi = *(const v8h*)(p + 16);
  return combine16(lo, hi);
}

// cross-lane reductions within each 16-lane half (rows of C/D layout stay inside a half)
__device__ __forceinline__ float half_max(float v) {
#pragma unroll
  for (int m = 8; m >= 1; m >>= 1) v = fmaxf(v, __shfl_xor(v, m, 32));
  return v;
}
__device__ __forceinline__ float half_sum(float v) {
#pragma unroll
  for (int m = 8; m >= 1; m >>= 1) v += __shfl_xor(v, m, 32);
  return v;
}

// ---------------- kernel 0: LDS-tiled transpose + f32->f16 convert ----------------
// W: K x N row-major (f32) -> Wt: N x K row-major (f16). 32x32 tile; both phases coalesced.
__global__ void transpose_cvt_kernel(const float* __restrict__ W, _Float16* __restrict__ Wt,
                                     int K, int N) {
  __shared__ float tile[32][33];           // +1 pad: no bank conflicts on column reads
  const int k0 = blockIdx.x * 32;
  const int n0 = blockIdx.y * 32;
  const int tx = threadIdx.x & 31;
  const int ty = threadIdx.x >> 5;         // 0..7
#pragma unroll
  for (int i = 0; i < 4; ++i) {
    int k = ty + 8 * i;
    tile[k][tx] = W[(size_t)(k0 + k) * N + n0 + tx];   // coalesced 128B rows
  }
  __syncthreads();
#pragma unroll
  for (int i = 0; i < 4; ++i) {
    int n = ty + 8 * i;
    Wt[(size_t)(n0 + n) * K + k0 + tx] = (_Float16)tile[tx][n];  // coalesced 64B rows
  }
}

// ---------------- kernel 1: LayerNorm -> f16 ----------------
__global__ void layernorm_kernel(const float* __restrict__ x,
                                 const float* __restrict__ gamma,
                                 const float* __restrict__ beta,
                                 _Float16* __restrict__ xn) {
  __shared__ float s1[256];
  __shared__ float s2[256];
  int row = blockIdx.x;
  const float* xr = x + (size_t)row * DIM;
  float a = 0.f, b = 0.f;
  for (int i = threadIdx.x; i < DIM; i += 256) { float v = xr[i]; a += v; b += v * v; }
  s1[threadIdx.x] = a; s2[threadIdx.x] = b;
  __syncthreads();
  for (int off = 128; off > 0; off >>= 1) {
    if (threadIdx.x < off) { s1[threadIdx.x] += s1[threadIdx.x + off];
                             s2[threadIdx.x] += s2[threadIdx.x + off]; }
    __syncthreads();
  }
  float mu   = s1[0] * (1.0f / DIM);
  float var  = s2[0] * (1.0f / DIM) - mu * mu;
  float rstd = rsqrtf(var + 1e-5f);
  for (int i = threadIdx.x; i < DIM; i += 256) {
    float v = (xr[i] - mu) * rstd * gamma[i] + beta[i];
    xn[(size_t)row * DIM + i] = (_Float16)v;
  }
}

// ---------------- kernel 2: QKV GEMM + head-split scatter ----------------
__device__ __forceinline__ void scatter_qkv(int grow, int gcol, _Float16 v,
                                            _Float16* __restrict__ qb,
                                            _Float16* __restrict__ kb,
                                            _Float16* __restrict__ vtb) {
  int b = grow >> 11;          // /SEQ
  int tok = grow & (SEQ - 1);
  if (gcol < DIM) {
    int h = gcol >> 6, d = gcol & 63;
    qb[(((size_t)(b * NH + h)) * SEQ + tok) * HD + d] = v;
  } else if (gcol < 2 * DIM) {
    int c = gcol - DIM; int h = c >> 6, d = c & 63;
    kb[(((size_t)(b * NH + h)) * SEQ + tok) * HD + d] = v;
  } else {
    int c = gcol - 2 * DIM; int h = c >> 6, d = c & 63;
    vtb[(((size_t)(b * NH + h)) * HD + d) * SEQ + tok] = v;  // V^T: (d, token)
  }
}

// Wave tile 32x64 (2x4 WMMA tiles, 8 wmma per k-step). Block: 8 waves = 4(M) x 2(N) -> 128x128.
__global__ void gemm_qkv_kernel(const _Float16* __restrict__ A,   // 4096x1024
                                const _Float16* __restrict__ Bt,  // 3072x1024 (w_qkv^T)
                                _Float16* __restrict__ qb,
                                _Float16* __restrict__ kb,
                                _Float16* __restrict__ vtb) {
  const int lane = threadIdx.x & 31;
  const int wave = threadIdx.x >> 5;
  const int wm = wave & 3, wn = wave >> 2;
  const int row0 = blockIdx.x * 128 + wm * 32;
  const int col0 = blockIdx.y * 128 + wn * 64;

  v8f acc[2][4] = {};
  for (int k0 = 0; k0 < DIM; k0 += 32) {
    v16h a0 = load_a_frag(A, DIM, row0,      k0, lane);
    v16h a1 = load_a_frag(A, DIM, row0 + 16, k0, lane);
#pragma unroll
    for (int j = 0; j < 4; ++j) {
      v16h bj = load_bt_frag(Bt, DIM, col0 + 16 * j, k0, lane);
      acc[0][j] = wmma_f16(a0, bj, acc[0][j]);
      acc[1][j] = wmma_f16(a1, bj, acc[1][j]);
    }
  }
  const int colb = lane & 15, rowb = (lane >> 4) * 8;
#pragma unroll
  for (int i = 0; i < 2; ++i)
#pragma unroll
    for (int j = 0; j < 4; ++j)
#pragma unroll
      for (int r = 0; r < 8; ++r)
        scatter_qkv(row0 + 16 * i + rowb + r, col0 + 16 * j + colb,
                    (_Float16)acc[i][j][r], qb, kb, vtb);
}

// ---------------- kernel 3: flash attention ----------------
// One wave per 16 q-rows per (b,h). Keys streamed 32 at a time, online softmax.
__global__ void attention_kernel(const _Float16* __restrict__ qb,
                                 const _Float16* __restrict__ kb,
                                 const _Float16* __restrict__ vtb,
                                 _Float16* __restrict__ attnb) {
  __shared__ __align__(16) _Float16 lds_p[8][16 * 32];   // per-wave P tile (16x32 f16)

  const int lane = threadIdx.x & 31;
  const int wave = threadIdx.x >> 5;
  const int wid  = blockIdx.x * 8 + wave;   // 0..4095
  const int qt   = wid & 127;               // q tile (2048/16)
  const int bh   = wid >> 7;                // 0..31
  const int b    = bh >> 4, h = bh & 15;
  const int qrow0 = qt * 16;

  const _Float16* qh  = qb  + (size_t)bh * SEQ * HD;
  const _Float16* kh  = kb  + (size_t)bh * SEQ * HD;
  const _Float16* vth = vtb + (size_t)bh * HD * SEQ;
  _Float16* lp = &lds_p[wave][0];

  const float scale = 0.125f;  // HD^-0.5

  // Q fragments (reused over all key tiles): contraction = head dim (64)
  v16h aq0 = load_a_frag(qh, HD, qrow0, 0,  lane);
  v16h aq1 = load_a_frag(qh, HD, qrow0, 32, lane);

  v8f acc0 = {}, acc1 = {}, acc2 = {}, acc3 = {};
  float Mrow[8], Lrow[8];
#pragma unroll
  for (int r = 0; r < 8; ++r) { Mrow[r] = -1e30f; Lrow[r] = 0.f; }

  const int colb = lane & 15, rowb = (lane >> 4) * 8;

  for (int t0 = 0; t0 < SEQ; t0 += 32) {
    // scores: S[q, key] = Q(16x64) @ K^T ; B[d, n] = K[t0+n, d] -> contiguous in K rows
    v16h bka0 = load_bt_frag(kh, HD, t0,      0,  lane);
    v16h bka1 = load_bt_frag(kh, HD, t0,      32, lane);
    v16h bkb0 = load_bt_frag(kh, HD, t0 + 16, 0,  lane);
    v16h bkb1 = load_bt_frag(kh, HD, t0 + 16, 32, lane);
    v8f z = {};
    v8f sa = wmma_f16(aq0, bka0, z); sa = wmma_f16(aq1, bka1, sa);
    v8f sb = wmma_f16(aq0, bkb0, z); sb = wmma_f16(aq1, bkb1, sb);

    float pa[8], pb[8];
#pragma unroll
    for (int r = 0; r < 8; ++r) {
      float x1 = sa[r] * scale, x2 = sb[r] * scale;
      float mx   = half_max(fmaxf(x1, x2));
      float newM = fmaxf(Mrow[r], mx);
      float f    = __expf(Mrow[r] - newM);
      float p1   = __expf(x1 - newM);
      float p2   = __expf(x2 - newM);
      float rs   = half_sum(p1 + p2);
      Lrow[r] = Lrow[r] * f + rs;
      Mrow[r] = newM;
      pa[r] = p1; pb[r] = p2;
      acc0[r] *= f; acc1[r] *= f; acc2[r] *= f; acc3[r] *= f;
    }

    // P (16x32 f16) -> private LDS slice, row-major
#pragma unroll
    for (int r = 0; r < 8; ++r) {
      lp[(rowb + r) * 32 + colb]      = (_Float16)pa[r];
      lp[(rowb + r) * 32 + 16 + colb] = (_Float16)pb[r];
    }
    // reload P in A-fragment layout (wave-private region; same-wave LDS ops are ordered)
    {
      int m = lane & 15, half = lane >> 4;
      v8h lo = *(const v8h*)(lp + m * 32 + half * 8);
      v8h hi = *(const v8h*)(lp + m * 32 + 16 + half * 8);
      v16h pf = combine16(lo, hi);

      // attn += P(16x32) @ V(32x64); B[k, n] = V[t0+k, d] = Vt[d, t0+k] contiguous
      v16h bv0 = load_bt_frag(vth, SEQ, 0,  t0, lane);
      v16h bv1 = load_bt_frag(vth, SEQ, 16, t0, lane);
      v16h bv2 = load_bt_frag(vth, SEQ, 32, t0, lane);
      v16h bv3 = load_bt_frag(vth, SEQ, 48, t0, lane);
      acc0 = wmma_f16(pf, bv0, acc0);
      acc1 = wmma_f16(pf, bv1, acc1);
      acc2 = wmma_f16(pf, bv2, acc2);
      acc3 = wmma_f16(pf, bv3, acc3);
    }
  }

  // normalize and store to (b*n, h*hd) f16 buffer
#pragma unroll
  for (int r = 0; r < 8; ++r) {
    float inv = 1.0f / Lrow[r];
    size_t rowoff = ((size_t)(b * SEQ + qrow0 + rowb + r)) * (NH * HD) + h * HD;
    attnb[rowoff +  0 + colb] = (_Float16)(acc0[r] * inv);
    attnb[rowoff + 16 + colb] = (_Float16)(acc1[r] * inv);
    attnb[rowoff + 32 + colb] = (_Float16)(acc2[r] * inv);
    attnb[rowoff + 48 + colb] = (_Float16)(acc3[r] * inv);
  }
}

// ---------------- kernel 4: output GEMM + bias -> fp32 ----------------
// Wave tile 32x64, block 128x128, same structure as QKV GEMM.
__global__ void gemm_out_kernel(const _Float16* __restrict__ A,   // 4096x1024 attn
                                const _Float16* __restrict__ Bt,  // 1024x1024 (w_out^T)
                                const float* __restrict__ bias,
                                float* __restrict__ out) {
  const int lane = threadIdx.x & 31;
  const int wave = threadIdx.x >> 5;
  const int wm = wave & 3, wn = wave >> 2;
  const int row0 = blockIdx.x * 128 + wm * 32;
  const int col0 = blockIdx.y * 128 + wn * 64;

  v8f acc[2][4] = {};
  for (int k0 = 0; k0 < DIM; k0 += 32) {
    v16h a0 = load_a_frag(A, DIM, row0,      k0, lane);
    v16h a1 = load_a_frag(A, DIM, row0 + 16, k0, lane);
#pragma unroll
    for (int j = 0; j < 4; ++j) {
      v16h bj = load_bt_frag(Bt, DIM, col0 + 16 * j, k0, lane);
      acc[0][j] = wmma_f16(a0, bj, acc[0][j]);
      acc[1][j] = wmma_f16(a1, bj, acc[1][j]);
    }
  }
  const int colb = lane & 15, rowb = (lane >> 4) * 8;
#pragma unroll
  for (int j = 0; j < 4; ++j) {
    float bia = bias[col0 + 16 * j + colb];
#pragma unroll
    for (int i = 0; i < 2; ++i)
#pragma unroll
      for (int r = 0; r < 8; ++r)
        out[(size_t)(row0 + 16 * i + rowb + r) * DIM + col0 + 16 * j + colb] =
            acc[i][j][r] + bia;
  }
}

// ---------------- host launcher ----------------
extern "C" void kernel_launch(void* const* d_in, const int* in_sizes, int n_in,
                              void* d_out, int out_size, void* d_ws, size_t ws_size,
                              hipStream_t stream) {
  const float* x     = (const float*)d_in[0];
  const float* gamma = (const float*)d_in[1];
  const float* beta  = (const float*)d_in[2];
  const float* w_qkv = (const float*)d_in[3];
  const float* w_out = (const float*)d_in[4];
  const float* b_out = (const float*)d_in[5];
  float* out = (float*)d_out;

  char* ws = (char*)d_ws;
  _Float16* xn     = (_Float16*)(ws + XN_OFF);
  _Float16* wqkvT  = (_Float16*)(ws + WQKVT_OFF);
  _Float16* woutT  = (_Float16*)(ws + WOUTT_OFF);
  _Float16* qb     = (_Float16*)(ws + QB_OFF);
  _Float16* kb     = (_Float16*)(ws + KB_OFF);
  _Float16* vtb    = (_Float16*)(ws + VTB_OFF);
  _Float16* attnb  = (_Float16*)(ws + ATTN_OFF);

  // weight transposes (f32 -> f16), LDS-tiled, fully coalesced
  transpose_cvt_kernel<<<dim3(DIM / 32, N_QKV / 32), 256, 0, stream>>>(w_qkv, wqkvT, DIM, N_QKV);
  transpose_cvt_kernel<<<dim3(DIM / 32, DIM / 32), 256, 0, stream>>>(w_out, woutT, DIM, DIM);

  // layernorm
  layernorm_kernel<<<ROWS, 256, 0, stream>>>(x, gamma, beta, xn);

  // QKV projection: block tile 128x128, wave tile 32x64
  gemm_qkv_kernel<<<dim3(ROWS / 128, N_QKV / 128), 256, 0, stream>>>(xn, wqkvT, qb, kb, vtb);

  // flash attention: 4096 waves, 8 per block
  attention_kernel<<<(BATCH * NH * (SEQ / 16)) / 8, 256, 0, stream>>>(qb, kb, vtb, attnb);

  // output projection + bias
  gemm_out_kernel<<<dim3(ROWS / 128, DIM / 128), 256, 0, stream>>>(attnb, woutT, b_out, out);
}
